// TemporalAttention_80564996539207
// MI455X (gfx1250) — matrix operand order repaired and checked by reading, not compiled, not run on hardware
//
#include <hip/hip_runtime.h>
#include <hip/hip_bf16.h>
#include <math.h>

#define Bdim 4
#define Sdim 2048
#define Ddim 1024
#define Hdim 16
#define HDdim 64

typedef _Float16 h16;
typedef __attribute__((ext_vector_type(16))) _Float16 v16h;
typedef __attribute__((ext_vector_type(8)))  _Float16 v8h;
typedef __attribute__((ext_vector_type(8)))  float    v8f;
typedef __attribute__((ext_vector_type(4)))  float    v4f;

static __device__ __forceinline__ v8f wmma32(v16h a, v16h b, v8f c) {
  // D = A(16x32 f16) * B(32x16 f16) + C(16x16 f32)
  return __builtin_amdgcn_wmma_f32_16x16x32_f16(
      /*neg_a=*/false, a, /*neg_b=*/false, b,
      /*c_mod=*/(short)0, c, /*reuse_a=*/false, /*reuse_b=*/false);
}

static __device__ __forceinline__ v16h pack16(v8h lo, v8h hi) {
  return __builtin_shufflevector(lo, hi, 0,1,2,3,4,5,6,7,8,9,10,11,12,13,14,15);
}

// Load the 16 fp32 values of an A-fragment (16 contiguous floats).
static __device__ __forceinline__ void loadA32(const float* p, v4f af[4]) {
  af[0] = *(const v4f*)(p);
  af[1] = *(const v4f*)(p + 4);
  af[2] = *(const v4f*)(p + 8);
  af[3] = *(const v4f*)(p + 12);
}

static __device__ __forceinline__ v16h cvtA(const v4f af[4]) {
  v16h r;
#pragma unroll
  for (int i = 0; i < 4; ++i) {
    r[i]      = (_Float16)af[0][i];
    r[4 + i]  = (_Float16)af[1][i];
    r[8 + i]  = (_Float16)af[2][i];
    r[12 + i] = (_Float16)af[3][i];
  }
  return r;
}

// ---------------------------------------------------------------------------
// Kernel 0: fp32 -> f16 elementwise cast (used for the weight matrices).
// ---------------------------------------------------------------------------
__global__ __launch_bounds__(256)
void ta_cvt_kernel(const float* __restrict__ src, h16* __restrict__ dst,
                   unsigned n) {
  unsigned i = (blockIdx.x * 256u + threadIdx.x) * 8u;
  if (i >= n) return;
  v4f a = *(const v4f*)(src + i);
  v4f b = *(const v4f*)(src + i + 4);
  v8h o;
#pragma unroll
  for (int j = 0; j < 4; ++j) { o[j] = (_Float16)a[j]; o[4 + j] = (_Float16)b[j]; }
  *(v8h*)(dst + i) = o;
}

// ---------------------------------------------------------------------------
// Kernel 1: fused QKV projection.  Y = X @ W^T + b, cast to f16.
//   grid = (M/64, N/128, 3)  block = 256 (8 waves: 4 M-tiles x 2 N-tiles)
//   sel 0 -> Qh[b,h,s,hd], sel 1 -> Kh[b,h,s,hd], sel 2 -> Vt[b,h,hd,s]
//   A converted fp32->f16 in flight; B pre-converted f16; software pipelined.
// ---------------------------------------------------------------------------
__global__ __launch_bounds__(256)
void ta_proj_kernel(const float* __restrict__ q, const float* __restrict__ k,
                    const float* __restrict__ v,
                    const h16* __restrict__ Wqh, const h16* __restrict__ Wkh,
                    const h16* __restrict__ Wvh,
                    const float* __restrict__ bq, const float* __restrict__ bk,
                    const float* __restrict__ bv,
                    h16* __restrict__ Qh, h16* __restrict__ Kh, h16* __restrict__ Vt)
{
  const int lane  = threadIdx.x & 31;
  const int wave  = threadIdx.x >> 5;
  const int wm    = wave & 3;
  const int wn    = wave >> 2;
  const int Mbase = blockIdx.x * 64 + wm * 16;
  const int Nbase = blockIdx.y * 128 + wn * 64;
  const int sel   = blockIdx.z;

  const float* X    = (sel == 0) ? q   : (sel == 1) ? k   : v;
  const h16*   W    = (sel == 0) ? Wqh : (sel == 1) ? Wkh : Wvh;
  const float* bias = (sel == 0) ? bq  : (sel == 1) ? bk  : bv;

  const int hl  = lane >> 4;
  const int l16 = lane & 15;

  const float* arow = X + (size_t)(Mbase + l16) * Ddim + (hl << 4);
  const h16* wrow[4];
#pragma unroll
  for (int nt = 0; nt < 4; ++nt)
    wrow[nt] = W + (size_t)(Nbase + nt * 16 + l16) * Ddim + (hl << 4);

  v8f acc[4] = {};
  v4f afc[4], afn[4];
  v16h bc[4], bn[4];

  loadA32(arow, afc);
#pragma unroll
  for (int nt = 0; nt < 4; ++nt) bc[nt] = *(const v16h*)(wrow[nt]);

#pragma unroll 2
  for (int kb = 0; kb < Ddim; kb += 32) {
    const int kn = (kb + 32 < Ddim) ? kb + 32 : kb;   // clamped prefetch
    loadA32(arow + kn, afn);
#pragma unroll
    for (int nt = 0; nt < 4; ++nt) bn[nt] = *(const v16h*)(wrow[nt] + kn);

    v16h A = cvtA(afc);
#pragma unroll
    for (int nt = 0; nt < 4; ++nt) acc[nt] = wmma32(A, bc[nt], acc[nt]);

#pragma unroll
    for (int i = 0; i < 4; ++i) afc[i] = afn[i];
#pragma unroll
    for (int nt = 0; nt < 4; ++nt) bc[nt] = bn[nt];
  }

#pragma unroll
  for (int nt = 0; nt < 4; ++nt) {
    const int n  = Nbase + nt * 16 + l16;
    const float bb = bias[n];
    const int h  = n >> 6;
    const int hd = n & 63;
#pragma unroll
    for (int r = 0; r < 8; ++r) {
      const int m = Mbase + r + (hl << 3);
      const int b = m >> 11;
      const int s = m & 2047;
      const float val = acc[nt][r] + bb;
      if (sel == 2) {
        Vt[(((size_t)b * Hdim + h) * HDdim + hd) * Sdim + s] = (h16)val;
      } else {
        h16* dst = (sel == 0) ? Qh : Kh;
        dst[(((size_t)b * Hdim + h) * Sdim + s) * HDdim + hd] = (h16)val;
      }
    }
  }
}

// ---------------------------------------------------------------------------
// Kernel 2: flash attention with additive exp(-gamma*dt) bias and mask.
//   grid = (B*H, S/128)  block = 256; each wave owns 16 query rows.
//   K/V fragments and td/mask loads issued at loop top so they overlap the
//   score WMMAs and the VALU-heavy online softmax.
// ---------------------------------------------------------------------------
__global__ __launch_bounds__(256)
void ta_attn_kernel(const h16* __restrict__ Qh, const h16* __restrict__ Kh,
                    const h16* __restrict__ Vt,
                    const float* __restrict__ td, const int* __restrict__ mask,
                    const float* __restrict__ gamma_p, h16* __restrict__ O)
{
  __shared__ h16 lds[8 * 16 * 32];   // 1 KB per wave
  const int lane = threadIdx.x & 31;
  const int wave = threadIdx.x >> 5;
  const int hl   = lane >> 4;
  const int l16  = lane & 15;
  const int bh   = blockIdx.x;            // b*H + h
  const int b    = bh >> 4;
  const int h    = bh & 15;
  const int q0   = blockIdx.y * 128 + wave * 16;
  const float gamma = gamma_p[0];

  h16* myl = lds + wave * (16 * 32);
  const int k0 = hl << 4;

  // Q A-fragments for hd 0..31 and 32..63 (persist across whole key loop)
  const h16* qrow = Qh + ((size_t)bh * Sdim + q0 + l16) * HDdim;
  const v16h aq0 = pack16(*(const v8h*)(qrow + k0),      *(const v8h*)(qrow + k0 + 8));
  const v16h aq1 = pack16(*(const v8h*)(qrow + 32 + k0), *(const v8h*)(qrow + 40 + k0));

  v8f acc[4] = {};
  float mrun[8], lrun[8];
#pragma unroll
  for (int r = 0; r < 8; ++r) { mrun[r] = -3.0e38f; lrun[r] = 0.0f; }

  const size_t tdb = (size_t)b * Sdim * Sdim;
  const h16* kbase = Kh + (size_t)bh * Sdim * HDdim + (hl << 4);
  const h16* vbase = Vt + (size_t)bh * HDdim * Sdim + (hl << 4);

#pragma unroll 1
  for (int kb = 0; kb < Sdim; kb += 32) {
    // ---- issue all global loads for this step up front ----
    v16h kf[4];   // K fragments: [tile t][hd-half]
#pragma unroll
    for (int t = 0; t < 2; ++t) {
      const h16* krow = kbase + (size_t)(kb + t * 16 + l16) * HDdim;
      kf[t * 2]     = *(const v16h*)(krow);
      kf[t * 2 + 1] = *(const v16h*)(krow + 32);
    }
    v16h vf[4];   // V fragments for hd tiles 0..3
#pragma unroll
    for (int j = 0; j < 4; ++j)
      vf[j] = *(const v16h*)(vbase + (size_t)(j * 16 + l16) * Sdim + kb);

    // td/mask folded into one bias value per element; masked -> -2e9
    float bias[2][8];
#pragma unroll
    for (int r = 0; r < 8; ++r) {
      const size_t rowoff = tdb + (size_t)(q0 + r + (hl << 3)) * Sdim + kb;
#pragma unroll
      for (int t = 0; t < 2; ++t) {
        const int key = t * 16 + l16;
        const float e = __expf(-gamma * td[rowoff + key]);
        bias[t][r] = (mask[rowoff + key] == 0) ? -2.0e9f : e;
      }
    }

    // ---- scores: S[16 x 32] as two 16x16 f32 tiles ----
    float p[2][8];
#pragma unroll
    for (int t = 0; t < 2; ++t) {
      v8f c = {};
      c = wmma32(aq0, kf[t * 2],     c);
      c = wmma32(aq1, kf[t * 2 + 1], c);
#pragma unroll
      for (int r = 0; r < 8; ++r) p[t][r] = c[r] * 0.125f + bias[t][r];
    }

    // ---- online softmax (row reductions stay inside a 16-lane half) ----
#pragma unroll
    for (int r = 0; r < 8; ++r) {
      float rm = fmaxf(p[0][r], p[1][r]);
#pragma unroll
      for (int off = 8; off >= 1; off >>= 1) rm = fmaxf(rm, __shfl_xor(rm, off, 32));
      const float mnew  = fmaxf(mrun[r], rm);
      const float scale = __expf(mrun[r] - mnew);
      mrun[r] = mnew;
      const float e0 = __expf(p[0][r] - mnew);
      const float e1 = __expf(p[1][r] - mnew);
      p[0][r] = e0; p[1][r] = e1;
      float rs = e0 + e1;
#pragma unroll
      for (int off = 8; off >= 1; off >>= 1) rs += __shfl_xor(rs, off, 32);
      lrun[r] = lrun[r] * scale + rs;
#pragma unroll
      for (int j = 0; j < 4; ++j) acc[j][r] *= scale;
    }

    // ---- P: C-layout -> LDS (row-major 16x32) -> A-layout ----
#pragma unroll
    for (int t = 0; t < 2; ++t)
#pragma unroll
      for (int r = 0; r < 8; ++r)
        myl[(r + (hl << 3)) * 32 + t * 16 + l16] = (h16)p[t][r];

    const v16h pa = pack16(*(const v8h*)(myl + l16 * 32 + k0),
                           *(const v8h*)(myl + l16 * 32 + k0 + 8));

    // ---- P @ V ----
#pragma unroll
    for (int j = 0; j < 4; ++j) acc[j] = wmma32(pa, vf[j], acc[j]);
  }

  // ---- normalize and store O[b,s,h*64+hd] as f16 ----
#pragma unroll
  for (int r = 0; r < 8; ++r) {
    const int qr = q0 + r + (hl << 3);
    const float inv = 1.0f / lrun[r];
    const size_t obase = ((size_t)b * Sdim + qr) * Ddim + h * HDdim;
#pragma unroll
    for (int j = 0; j < 4; ++j)
      O[obase + j * 16 + l16] = (h16)(acc[j][r] * inv);
  }
}

// ---------------------------------------------------------------------------
// Kernel 3: output projection.  out = O @ Wo^T + bo  (fp32 result)
//   All-f16 operands, software pipelined.
// ---------------------------------------------------------------------------
__global__ __launch_bounds__(256)
void ta_outproj_kernel(const h16* __restrict__ O, const h16* __restrict__ Woh,
                       const float* __restrict__ bo, float* __restrict__ out)
{
  const int lane  = threadIdx.x & 31;
  const int wave  = threadIdx.x >> 5;
  const int wm    = wave & 3;
  const int wn    = wave >> 2;
  const int Mbase = blockIdx.x * 64 + wm * 16;
  const int Nbase = blockIdx.y * 128 + wn * 64;
  const int hl    = lane >> 4;
  const int l16   = lane & 15;
  const int k0    = hl << 4;

  const h16* arow = O + (size_t)(Mbase + l16) * Ddim;
  const h16* wrow[4];
#pragma unroll
  for (int nt = 0; nt < 4; ++nt)
    wrow[nt] = Woh + (size_t)(Nbase + nt * 16 + l16) * Ddim + (hl << 4);

  v8f acc[4] = {};
  v8h alo, ahi, alon, ahin;
  v16h bc[4], bn[4];

  alo = *(const v8h*)(arow + k0);
  ahi = *(const v8h*)(arow + k0 + 8);
#pragma unroll
  for (int nt = 0; nt < 4; ++nt) bc[nt] = *(const v16h*)(wrow[nt]);

#pragma unroll 2
  for (int kb = 0; kb < Ddim; kb += 32) {
    const int kn = (kb + 32 < Ddim) ? kb + 32 : kb;
    alon = *(const v8h*)(arow + kn + k0);
    ahin = *(const v8h*)(arow + kn + k0 + 8);
#pragma unroll
    for (int nt = 0; nt < 4; ++nt) bn[nt] = *(const v16h*)(wrow[nt] + kn);

    v16h A = pack16(alo, ahi);
#pragma unroll
    for (int nt = 0; nt < 4; ++nt) acc[nt] = wmma32(A, bc[nt], acc[nt]);

    alo = alon; ahi = ahin;
#pragma unroll
    for (int nt = 0; nt < 4; ++nt) bc[nt] = bn[nt];
  }

#pragma unroll
  for (int nt = 0; nt < 4; ++nt) {
    const int n = Nbase + nt * 16 + l16;
    const float bb = bo[n];
#pragma unroll
    for (int r = 0; r < 8; ++r) {
      const int m = Mbase + r + (hl << 3);
      out[(size_t)m * Ddim + n] = acc[nt][r] + bb;
    }
  }
}

// ---------------------------------------------------------------------------
extern "C" void kernel_launch(void* const* d_in, const int* in_sizes, int n_in,
                              void* d_out, int out_size, void* d_ws, size_t ws_size,
                              hipStream_t stream) {
  const float* q   = (const float*)d_in[0];
  const float* k   = (const float*)d_in[1];
  const float* v   = (const float*)d_in[2];
  const float* td  = (const float*)d_in[3];
  const int*   msk = (const int*)d_in[4];
  const float* Wq  = (const float*)d_in[5];
  const float* bq  = (const float*)d_in[6];
  const float* Wk  = (const float*)d_in[7];
  const float* bk  = (const float*)d_in[8];
  const float* Wv  = (const float*)d_in[9];
  const float* bv  = (const float*)d_in[10];
  const float* Wo  = (const float*)d_in[11];
  const float* bo  = (const float*)d_in[12];
  const float* gam = (const float*)d_in[13];
  float* out = (float*)d_out;

  const size_t elems = (size_t)Bdim * Hdim * Sdim * HDdim;  // == B*S*D
  const size_t welems = (size_t)Ddim * Ddim;
  h16* Qh  = (h16*)d_ws;
  h16* Kh  = Qh + elems;
  h16* Vt  = Kh + elems;
  h16* O   = Vt + elems;
  h16* Wqh = O + elems;
  h16* Wkh = Wqh + welems;
  h16* Wvh = Wkh + welems;
  h16* Woh = Wvh + welems;

  dim3 blk(256);

  // weight casts: 1M elements each, 8 per thread
  const unsigned wN = (unsigned)welems;
  const unsigned cvtBlocks = wN / (256u * 8u);
  ta_cvt_kernel<<<cvtBlocks, blk, 0, stream>>>(Wq, Wqh, wN);
  ta_cvt_kernel<<<cvtBlocks, blk, 0, stream>>>(Wk, Wkh, wN);
  ta_cvt_kernel<<<cvtBlocks, blk, 0, stream>>>(Wv, Wvh, wN);
  ta_cvt_kernel<<<cvtBlocks, blk, 0, stream>>>(Wo, Woh, wN);

  // fused QKV projection: M=B*S=8192 (64/block), N=1024 (128/block), z = {Q,K,V}
  ta_proj_kernel<<<dim3((Bdim * Sdim) / 64, Ddim / 128, 3), blk, 0, stream>>>(
      q, k, v, Wqh, Wkh, Wvh, bq, bk, bv, Qh, Kh, Vt);

  // flash attention: (B*H) x (S/128) blocks, 8 waves x 16 query rows
  ta_attn_kernel<<<dim3(Bdim * Hdim, Sdim / 128), blk, 0, stream>>>(
      Qh, Kh, Vt, td, msk, gam, O);

  // output projection
  ta_outproj_kernel<<<dim3((Bdim * Sdim) / 64, Ddim / 128), blk, 0, stream>>>(
      O, Woh, bo, out);
}
